// CausalPrefixMemory_72524817760546
// MI455X (gfx1250) — matrix operand order repaired and checked by reading, compiled, not run on hardware
//
#include <hip/hip_runtime.h>
#include <hip/hip_bf16.h>
#include <math.h>
#include <stdint.h>

// ---------------------------------------------------------------------------
// CausalPrefixMemory on MI455X (gfx1250, wave32, WMMA + async Global->LDS)
//   xb,W*b = bf16 copies (one-time convert; GEMM loops stay VALU-free)
//   P      = sigmoid(xb Wgb^T) * (xb Wub^T)   [bf16 WMMA, 128x32 block tile,
//                                              weights async-staged in LDS,
//                                              64-K double-buffered stages]
//   M      = cumsum_S(P)                       [3-pass parallel scan]
//   Mb     = bf16( alpha * M )                 [fused RMSNorm scale]
//   out    = Mb Wob^T                          [bf16 WMMA, async LDS weights]
// ---------------------------------------------------------------------------

typedef __attribute__((ext_vector_type(16))) __bf16 v16bf;
typedef __attribute__((ext_vector_type(8)))  __bf16 v8bf;
typedef __attribute__((ext_vector_type(8)))  float  v8f;

constexpr int B  = 4;
constexpr int S  = 8192;
constexpr int D  = 1024;
constexpr int SD = 512;
constexpr int M_ROWS = B * S;          // 32768 flattened rows
constexpr float RMS_EPS = 1.1920929e-07f;
constexpr int CHUNK = 256;             // scan chunk along S
constexpr int NCH   = S / CHUNK;       // 32 chunks

union FragU { v16bf v; v8bf h[2]; };

// A-fragment (16x32 bf16): lane holds row m=lane&15, ko=(lane<16)?0:8
//   elems 0..7 at K=ko+i, elems 8..15 at K=16+ko+i  -> gap 16
__device__ inline v16bf frag_ld(const __bf16* __restrict__ p, int gap) {
  FragU u;
  u.h[0] = *(const v8bf*)(p);
  u.h[1] = *(const v8bf*)(p + gap);
  return u.v;
}

// B-fragment read out of an LDS subtile laid out as [32 cols][32 k] bf16.
// lane holds col n, kb=(lane<16)?0:8; elems 0..7 at K=kb+i, 8..15 at K=16+kb+i (two ds_load_b128).
__device__ inline v16bf frag_b_lds(const __bf16* tile, int n, int kb) {
  FragU u;
  const v8bf* p = (const v8bf*)(tile + n * 32 + kb);
  u.h[0] = p[0];
  u.h[1] = p[2];
  return u.v;
}

// Async-stage one 32(col) x 32(k) bf16 weight subtile (2 KB) into LDS:
// 128 threads x one global_load_async_to_lds_b128 (16 B/lane, ASYNCcnt).
__device__ inline void stage_tile_async(const __bf16* __restrict__ W,
                                        __bf16* tile, int col0, int kk,
                                        int stride, int tid) {
  const int n  = tid >> 2;             // 0..31 : weight row (output col)
  const int kp = (tid & 3) * 8;        // bf16 offset within 32-k slice
  const __bf16* g = W + (size_t)(col0 + n) * stride + kk + kp;
  const uint32_t l = (uint32_t)(size_t)(tile + n * 32 + kp);
  asm volatile("global_load_async_to_lds_b128 %0, %1, off"
               :: "v"(l), "v"((unsigned long long)(size_t)g)
               : "memory");
}

__device__ inline void wait_async_all() {
  asm volatile("s_wait_asynccnt 0x0" ::: "memory");
}

#define WMMA_BF16(A, Bf, C) \
  __builtin_amdgcn_wmma_f32_16x16x32_bf16(false, (A), false, (Bf), (short)0, (C), false, false)

// ---- one-time f32 -> bf16 convert (8 elems / thread) ----------------------
__global__ __launch_bounds__(256) void k_cvt_bf16(
    const float* __restrict__ src, __bf16* __restrict__ dst, int n8) {
  const int i = blockIdx.x * 256 + threadIdx.x;
  if (i >= n8) return;
  const float4* s = (const float4*)src + (size_t)i * 2;
  float4 a = s[0], b = s[1];
  v8bf o;
  o[0] = (__bf16)a.x; o[1] = (__bf16)a.y; o[2] = (__bf16)a.z; o[3] = (__bf16)a.w;
  o[4] = (__bf16)b.x; o[5] = (__bf16)b.y; o[6] = (__bf16)b.z; o[7] = (__bf16)b.w;
  ((v8bf*)dst)[i] = o;
}

// ---- kernel 1: P = sigmoid(xb Wgb^T) * (xb Wub^T) -------------------------
// block = 128(M) x 32(N); 4 waves stacked in M share LDS weight tiles.
// 64-K stages (2 subtiles) double-buffered: 16 WMMAs per barrier per wave.
__global__ __launch_bounds__(128) void k_gemm_gate_update(
    const __bf16* __restrict__ xb, const __bf16* __restrict__ Wub,
    const __bf16* __restrict__ Wgb, float* __restrict__ P) {
  __shared__ __align__(16) __bf16 sg[2][2][32 * 32];  // gate:   2 buf x 2 sub
  __shared__ __align__(16) __bf16 su[2][2][32 * 32];  // update: 2 buf x 2 sub
  const int tid  = threadIdx.x;
  const int lane = tid & 31;
  const int wave = tid >> 5;
  const int NB = SD / 32;                            // 16 N-blocks
  const int mb = blockIdx.x / NB, nb = blockIdx.x % NB;
  const int row0 = mb * 128 + wave * 32;
  const int col0 = nb * 32;
  const int m  = lane & 15;
  const int ko = (lane < 16) ? 0 : 8;
  const int kb = (lane < 16) ? 0 : 8;

  const __bf16* xr0 = xb + (size_t)(row0 + m) * D;
  const __bf16* xr1 = xr0 + (size_t)16 * D;

  stage_tile_async(Wgb, sg[0][0], col0, 0, D, tid);
  stage_tile_async(Wgb, sg[0][1], col0, 32, D, tid);
  stage_tile_async(Wub, su[0][0], col0, 0, D, tid);
  stage_tile_async(Wub, su[0][1], col0, 32, D, tid);
  wait_async_all();
  __syncthreads();

  v8f aG[2][2] = {};
  v8f aU[2][2] = {};
  int cur = 0;
  for (int kk = 0; kk < D; kk += 64, cur ^= 1) {
    if (kk + 64 < D) {                               // prefetch next 64-K tile
      stage_tile_async(Wgb, sg[cur ^ 1][0], col0, kk + 64, D, tid);
      stage_tile_async(Wgb, sg[cur ^ 1][1], col0, kk + 96, D, tid);
      stage_tile_async(Wub, su[cur ^ 1][0], col0, kk + 64, D, tid);
      stage_tile_async(Wub, su[cur ^ 1][1], col0, kk + 96, D, tid);
    }
    __builtin_prefetch(xr0 + kk + 512, 0, 3);        // global_prefetch_b8
    __builtin_prefetch(xr1 + kk + 512, 0, 3);
#pragma unroll
    for (int t = 0; t < 2; ++t) {
      const int kks = kk + t * 32;
      v16bf a0 = frag_ld(xr0 + kks + ko, 16);
      v16bf a1 = frag_ld(xr1 + kks + ko, 16);
      v16bf g0 = frag_b_lds(sg[cur][t], m, kb);
      v16bf g1 = frag_b_lds(sg[cur][t], 16 + m, kb);
      v16bf u0 = frag_b_lds(su[cur][t], m, kb);
      v16bf u1 = frag_b_lds(su[cur][t], 16 + m, kb);
      aG[0][0] = WMMA_BF16(a0, g0, aG[0][0]);
      aG[0][1] = WMMA_BF16(a0, g1, aG[0][1]);
      aG[1][0] = WMMA_BF16(a1, g0, aG[1][0]);
      aG[1][1] = WMMA_BF16(a1, g1, aG[1][1]);
      aU[0][0] = WMMA_BF16(a0, u0, aU[0][0]);
      aU[0][1] = WMMA_BF16(a0, u1, aU[0][1]);
      aU[1][0] = WMMA_BF16(a1, u0, aU[1][0]);
      aU[1][1] = WMMA_BF16(a1, u1, aU[1][1]);
    }
    wait_async_all();                                // next tile landed
    __syncthreads();                                 // all readers done
  }

  // C/D layout: elem r -> row = r + 8*(lane>=16), col = lane&15
  const int rbase = (lane >> 4) << 3;
#pragma unroll
  for (int i = 0; i < 2; ++i)
#pragma unroll
    for (int j = 0; j < 2; ++j)
#pragma unroll
      for (int r = 0; r < 8; ++r) {
        float g = aG[i][j][r];
        float u = aU[i][j][r];
        float p = u * (1.0f / (1.0f + __expf(-g)));
        P[(size_t)(row0 + i * 16 + rbase + r) * SD + col0 + j * 16 + m] = p;
      }
}

// ---- scan pass A: per-chunk sums along S ----------------------------------
__global__ __launch_bounds__(256) void k_chunk_sums(
    const float* __restrict__ P, float* __restrict__ cs) {
  const int KG = SD / 256;
  const int kg = blockIdx.x % KG;
  const int ch = (blockIdx.x / KG) % NCH;
  const int b  = blockIdx.x / (KG * NCH);
  const int k  = kg * 256 + threadIdx.x;
  const float* p = P + ((size_t)b * S + (size_t)ch * CHUNK) * SD + k;
  float sum = 0.0f;
#pragma unroll 8
  for (int s = 0; s < CHUNK; ++s) sum += p[(size_t)s * SD];
  cs[((size_t)b * SD + k) * NCH + ch] = sum;
}

// ---- scan pass B: exclusive scan of chunk sums ----------------------------
__global__ __launch_bounds__(256) void k_scan_offsets(float* __restrict__ cs) {
  const int idx = blockIdx.x * 256 + threadIdx.x;
  if (idx >= B * SD) return;
  float* p = cs + (size_t)idx * NCH;
  float run = 0.0f;
#pragma unroll
  for (int c = 0; c < NCH; ++c) { float v = p[c]; p[c] = run; run += v; }
}

// ---- scan pass C: in-place cumsum with chunk carry ------------------------
__global__ __launch_bounds__(256) void k_cumsum(
    float* __restrict__ P, const float* __restrict__ cs) {
  const int KG = SD / 256;
  const int kg = blockIdx.x % KG;
  const int ch = (blockIdx.x / KG) % NCH;
  const int b  = blockIdx.x / (KG * NCH);
  const int k  = kg * 256 + threadIdx.x;
  float* p = P + ((size_t)b * S + (size_t)ch * CHUNK) * SD + k;
  float carry = cs[((size_t)b * SD + k) * NCH + ch];
#pragma unroll 8
  for (int s = 0; s < CHUNK; ++s) {
    carry += p[(size_t)s * SD];
    p[(size_t)s * SD] = carry;
  }
}

// ---- fused RMS scale + bf16 convert: Mb = bf16(alpha * M) -----------------
__global__ __launch_bounds__(256) void k_rms_scale_cvt(
    const float* __restrict__ Mm, __bf16* __restrict__ Mb) {
  const int lane = threadIdx.x & 31;
  const int wave = threadIdx.x >> 5;
  const int row  = blockIdx.x * 8 + wave;
  const float* mr = Mm + (size_t)row * SD + lane * 16;
  float v[16];
  const float4* p = (const float4*)mr;
  float4 q0 = p[0], q1 = p[1], q2 = p[2], q3 = p[3];
  v[0]=q0.x; v[1]=q0.y; v[2]=q0.z; v[3]=q0.w;
  v[4]=q1.x; v[5]=q1.y; v[6]=q1.z; v[7]=q1.w;
  v[8]=q2.x; v[9]=q2.y; v[10]=q2.z; v[11]=q2.w;
  v[12]=q3.x; v[13]=q3.y; v[14]=q3.z; v[15]=q3.w;
  float ss = 0.0f;
#pragma unroll
  for (int i = 0; i < 16; ++i) ss += v[i] * v[i];
#pragma unroll
  for (int off = 16; off > 0; off >>= 1) ss += __shfl_xor(ss, off, 32);
  const int s = row & (S - 1);
  const float denom = (float)(s + 1);
  const float alpha = rsqrtf(ss / (denom * denom * (float)SD) + RMS_EPS) / denom;
  v8bf o0, o1;
#pragma unroll
  for (int i = 0; i < 8; ++i) {
    o0[i] = (__bf16)(v[i] * alpha);
    o1[i] = (__bf16)(v[8 + i] * alpha);
  }
  v8bf* d = (v8bf*)(Mb + (size_t)row * SD + lane * 16);
  d[0] = o0;
  d[1] = o1;
}

// ---- kernel 4: out = Mb Wob^T ---------------------------------------------
// Same async-LDS weight staging; 64-K double-buffered stages.
__global__ __launch_bounds__(128) void k_gemm_out(
    const __bf16* __restrict__ Mb, const __bf16* __restrict__ Wob,
    float* __restrict__ out) {
  __shared__ __align__(16) __bf16 so[2][2][32 * 32];
  const int tid  = threadIdx.x;
  const int lane = tid & 31;
  const int wave = tid >> 5;
  const int NB = D / 32;                             // 32 N-blocks
  const int mb = blockIdx.x / NB, nb = blockIdx.x % NB;
  const int row0 = mb * 128 + wave * 32;
  const int col0 = nb * 32;
  const int m  = lane & 15;
  const int ko = (lane < 16) ? 0 : 8;
  const int kb = (lane < 16) ? 0 : 8;

  const __bf16* mr0 = Mb + (size_t)(row0 + m) * SD;
  const __bf16* mr1 = mr0 + (size_t)16 * SD;

  stage_tile_async(Wob, so[0][0], col0, 0, SD, tid);
  stage_tile_async(Wob, so[0][1], col0, 32, SD, tid);
  wait_async_all();
  __syncthreads();

  v8f acc[2][2] = {};
  int cur = 0;
  for (int kk = 0; kk < SD; kk += 64, cur ^= 1) {
    if (kk + 64 < SD) {
      stage_tile_async(Wob, so[cur ^ 1][0], col0, kk + 64, SD, tid);
      stage_tile_async(Wob, so[cur ^ 1][1], col0, kk + 96, SD, tid);
    }
    __builtin_prefetch(mr0 + kk + 256, 0, 3);
    __builtin_prefetch(mr1 + kk + 256, 0, 3);
#pragma unroll
    for (int t = 0; t < 2; ++t) {
      const int kks = kk + t * 32;
      v16bf a0 = frag_ld(mr0 + kks + ko, 16);
      v16bf a1 = frag_ld(mr1 + kks + ko, 16);
      v16bf b0 = frag_b_lds(so[cur][t], m, kb);
      v16bf b1 = frag_b_lds(so[cur][t], 16 + m, kb);
      acc[0][0] = WMMA_BF16(a0, b0, acc[0][0]);
      acc[0][1] = WMMA_BF16(a0, b1, acc[0][1]);
      acc[1][0] = WMMA_BF16(a1, b0, acc[1][0]);
      acc[1][1] = WMMA_BF16(a1, b1, acc[1][1]);
    }
    wait_async_all();
    __syncthreads();
  }

  const int rbase = (lane >> 4) << 3;
#pragma unroll
  for (int i = 0; i < 2; ++i)
#pragma unroll
    for (int j = 0; j < 2; ++j)
#pragma unroll
      for (int r = 0; r < 8; ++r)
        out[(size_t)(row0 + i * 16 + rbase + r) * D + col0 + j * 16 + m] =
            acc[i][j][r];
}

// ---------------------------------------------------------------------------
extern "C" void kernel_launch(void* const* d_in, const int* in_sizes, int n_in,
                              void* d_out, int out_size, void* d_ws, size_t ws_size,
                              hipStream_t stream) {
  const float* x  = (const float*)d_in[0];
  const float* Wu = (const float*)d_in[1];
  const float* Wg = (const float*)d_in[2];
  const float* Wo = (const float*)d_in[3];
  float* out = (float*)d_out;

  // workspace layout:
  //   [0,   64M)  xb (bf16 x) -> reused as Mb (bf16, 32 MB) after GEMM1
  //   [64M, 128M) P  (f32 gate*upd, then cumsum M in place)
  //   [128M, ..)  Wgb | Wub | Wob (bf16, 1 MB each), cs (256 KB)
  char* ws = (char*)d_ws;
  __bf16* xb  = (__bf16*)ws;
  __bf16* Mb  = (__bf16*)ws;                                   // aliases xb
  float*  P   = (float*)(ws + (size_t)M_ROWS * D * 2);
  char*   w2  = ws + (size_t)M_ROWS * D * 2 + (size_t)M_ROWS * SD * 4;
  __bf16* Wgb = (__bf16*)w2;
  __bf16* Wub = (__bf16*)(w2 + (size_t)SD * D * 2);
  __bf16* Wob = (__bf16*)(w2 + (size_t)SD * D * 4);
  float*  cs  = (float*)(w2 + (size_t)SD * D * 6);

  // 0) one-time bf16 conversions
  k_cvt_bf16<<<(M_ROWS * D / 8) / 256, 256, 0, stream>>>(x, xb, M_ROWS * D / 8);
  k_cvt_bf16<<<(SD * D / 8) / 256, 256, 0, stream>>>(Wg, Wgb, SD * D / 8);
  k_cvt_bf16<<<(SD * D / 8) / 256, 256, 0, stream>>>(Wu, Wub, SD * D / 8);
  k_cvt_bf16<<<(SD * D / 8) / 256, 256, 0, stream>>>(Wo, Wob, SD * D / 8);

  // 1) gate/update GEMMs fused with sigmoid*upd (128x32 block, async LDS B)
  k_gemm_gate_update<<<(M_ROWS / 128) * (SD / 32), 128, 0, stream>>>(
      xb, Wub, Wgb, P);

  // 2) parallel scan along S
  const int scan_blocks = B * NCH * (SD / 256);
  k_chunk_sums<<<scan_blocks, 256, 0, stream>>>(P, cs);
  k_scan_offsets<<<(B * SD + 255) / 256, 256, 0, stream>>>(cs);
  k_cumsum<<<scan_blocks, 256, 0, stream>>>(P, cs);

  // 3) fused RMSNorm scale + bf16 convert (Mb overwrites dead xb region)
  k_rms_scale_cvt<<<M_ROWS / 8, 256, 0, stream>>>(P, Mb);

  // 4) output GEMM (128x32 block, async LDS B)
  k_gemm_out<<<(M_ROWS / 128) * (D / 32), 128, 0, stream>>>(Mb, Wob, out);
}
